// AttentionBlock_66425964200113
// MI455X (gfx1250) — matrix operand order repaired and checked by reading, not compiled, not run on hardware
//
#include <hip/hip_runtime.h>
#include <hip/hip_bf16.h>
#include <math.h>

// Problem constants (reference: N, T, C, K, V = 32, 1024, 512, 512, 512)
#define NB   32
#define TDIM 1024
#define CDIM 512
#define KDIM 512
#define VDIM 512

typedef __bf16 bf16_t;
typedef bf16_t v16bf __attribute__((ext_vector_type(16)));
typedef bf16_t v8bf  __attribute__((ext_vector_type(8)));
typedef bf16_t v2bf  __attribute__((ext_vector_type(2)));
typedef float  v8f   __attribute__((ext_vector_type(8)));

#define MODE_PROJ   0
#define MODE_LOGITS 1
#define MODE_AV     2

// ---------------------------------------------------------------------------
// CDNA5 async global->LDS copy (ASYNCcnt-tracked DMA, bypasses VGPRs).
// lds_addr: wave-relative LDS byte address = low 32 bits of the generic
// pointer (the LDS aperture occupies bits [63:32] per the ISA aperture map).
// ---------------------------------------------------------------------------
__device__ __forceinline__ void async_ld16(unsigned lds_addr, const void* gaddr) {
    asm volatile("global_load_async_to_lds_b128 %0, %1, off"
                 :: "v"(lds_addr), "v"(gaddr)
                 : "memory");
}
__device__ __forceinline__ void wait_async_all() {
    asm volatile("s_wait_asynccnt 0x0" ::: "memory");
}
__device__ __forceinline__ unsigned lds_off(const void* p) {
    return (unsigned)(unsigned long long)p;
}

// ---------------------------------------------------------------------------
// f32 -> bf16 elementwise convert (8 elems / thread, 16B stores)
// ---------------------------------------------------------------------------
__global__ void cvt_f32_to_bf16(const float* __restrict__ src,
                                bf16_t* __restrict__ dst, long long n) {
    long long i = ((long long)blockIdx.x * blockDim.x + threadIdx.x) * 8;
    if (i >= n) return;
    float4 a = *reinterpret_cast<const float4*>(src + i);
    float4 b = *reinterpret_cast<const float4*>(src + i + 4);
    v8bf o;
    o[0] = (bf16_t)a.x; o[1] = (bf16_t)a.y; o[2] = (bf16_t)a.z; o[3] = (bf16_t)a.w;
    o[4] = (bf16_t)b.x; o[5] = (bf16_t)b.y; o[6] = (bf16_t)b.z; o[7] = (bf16_t)b.w;
    *reinterpret_cast<v8bf*>(dst + i) = o;
}

// ---------------------------------------------------------------------------
// Tiled bf16 WMMA GEMM with double-buffered LDS and async global->LDS staging.
//   BM=BN=128, BK=32, 256 threads = 8 waves; wave grid 4(M) x 2(N);
//   each wave: 2x4 tiles of v_wmma_f32_16x16x32_bf16 (32x64 output).
// MODE_PROJ  : B is [K][N] row-major (weights); out = bf16(acc + bias[n])
// MODE_LOGITS: B is [N][K] row-major (K-matrix rows, async path); out f32 =
//              scale*acc, masked -inf where col(s) > row(t)
// MODE_AV    : B is [K][N] row-major (V-matrix); out f32 into concat region
// All dims divide the tile sizes exactly for this problem -> no bounds checks.
// ---------------------------------------------------------------------------
template <int MODE>
__launch_bounds__(256, 2)
__global__ void gemm_wmma(const bf16_t* __restrict__ Abase, long long lda, long long aBatch,
                          const bf16_t* __restrict__ Bbase, long long ldb, long long bBatch,
                          float* __restrict__ outF, bf16_t* __restrict__ outB,
                          long long ldo, long long oBatch,
                          const float* __restrict__ bias,
                          int Ksize, float scale) {
    constexpr int BM = 128, BN = 128, BK = 32, LDSW = 40; // 40-elem pitch = 80B = 5*16B
    __shared__ bf16_t As[2][BM * LDSW];
    __shared__ bf16_t Bs[2][BN * LDSW];

    const int tid   = threadIdx.x;
    const int lane  = tid & 31;
    const int wave  = tid >> 5;
    const int waveM = wave >> 1;   // 0..3
    const int waveN = wave & 1;    // 0..1
    const int l16   = lane & 15;
    const int lHi   = lane >> 4;   // 0 or 1

    const int z = blockIdx.z;
    const bf16_t* A = Abase + (size_t)z * aBatch;
    const bf16_t* B = Bbase + (size_t)z * bBatch;

    const long long mBase = (long long)blockIdx.y * BM;
    const long long nBase = (long long)blockIdx.x * BN;

    // staging roles
    const int r   = tid >> 1, h = tid & 1;     // A / direct-B: 32B per thread
    const int p   = tid >> 4;                  // transpose-B: k-pair 0..15
    const int n0t = (tid & 15) * 8;            // transpose-B: 8 columns

    // A tile: rows contiguous in k -> two 16B async DMAs per thread
    auto stageA_async = [&](int k0, int buf) {
        const bf16_t* g = A + (mBase + r) * lda + k0 + h * 16;
        unsigned l = lds_off(&As[buf][r * LDSW + h * 16]);
        async_ld16(l, g);
        async_ld16(l + 16, g + 8);
    };
    // direct B tile ([n][k] source, LOGITS): same shape as A staging
    auto stageB_async = [&](int k0, int buf) {
        const bf16_t* g = B + (nBase + r) * ldb + k0 + h * 16;
        unsigned l = lds_off(&Bs[buf][r * LDSW + h * 16]);
        async_ld16(l, g);
        async_ld16(l + 16, g + 8);
    };
    // transpose B tile ([k][n] source, PROJ/AV): regs now, LDS after compute
    auto loadBrows = [&](int k0, uint4& u0, uint4& u1) {
        const bf16_t* r0 = B + (long long)(k0 + 2 * p) * ldb + nBase + n0t;
        u0 = *reinterpret_cast<const uint4*>(r0);
        u1 = *reinterpret_cast<const uint4*>(r0 + ldb);
    };
    auto storeBT = [&](int buf, uint4 u0, uint4 u1) {
        const bf16_t* e0 = reinterpret_cast<const bf16_t*>(&u0);
        const bf16_t* e1 = reinterpret_cast<const bf16_t*>(&u1);
#pragma unroll
        for (int j = 0; j < 8; ++j) {
            v2bf w; w.x = e0[j]; w.y = e1[j];
            *reinterpret_cast<v2bf*>(&Bs[buf][(n0t + j) * LDSW + 2 * p]) = w;
        }
    };

    v8f zero = {0.f, 0.f, 0.f, 0.f, 0.f, 0.f, 0.f, 0.f};
    v8f acc[2][4];
#pragma unroll
    for (int i = 0; i < 2; ++i)
#pragma unroll
        for (int j = 0; j < 4; ++j) acc[i][j] = zero;

    const int nk = Ksize / BK;

    // ---- prologue: stage tile 0 into buffer 0 ----
    stageA_async(0, 0);
    if (MODE == MODE_LOGITS) {
        stageB_async(0, 0);
    } else {
        uint4 u0, u1;
        loadBrows(0, u0, u1);
        storeBT(0, u0, u1);
    }
    wait_async_all();
    __syncthreads();

    for (int ks = 0; ks < nk; ++ks) {
        const int  cur  = ks & 1, nxt = cur ^ 1;
        const bool more = (ks + 1) < nk;

        // ---- kick off staging of the NEXT K-slab (overlaps with WMMA) ----
        uint4 u0, u1;
        if (more) {
            stageA_async((ks + 1) * BK, nxt);
            if (MODE == MODE_LOGITS) stageB_async((ks + 1) * BK, nxt);
            else                     loadBrows((ks + 1) * BK, u0, u1);
        }

        // ---- fragments per documented 16-bit A/B layouts ----
        // A lane L: row = L&15; elems 0..7 -> k=(L>>4)*8+0..7, elems 8..15 -> +16
        const bf16_t* Ab = As[cur];
        const bf16_t* Bb = Bs[cur];
        v16bf afrag[2];
#pragma unroll
        for (int tm = 0; tm < 2; ++tm) {
            const int m = waveM * 32 + tm * 16 + l16;
            v8bf a0 = *reinterpret_cast<const v8bf*>(&Ab[m * LDSW + lHi * 8]);
            v8bf a1 = *reinterpret_cast<const v8bf*>(&Ab[m * LDSW + 16 + lHi * 8]);
            afrag[tm] = __builtin_shufflevector(a0, a1, 0, 1, 2, 3, 4, 5, 6, 7,
                                                8, 9, 10, 11, 12, 13, 14, 15);
        }
        // B lane L: col = L&15; elems 0..7 -> k=(L>>4)*8+0..7, elems 8..15 -> +16
        v16bf bfrag[4];
#pragma unroll
        for (int tn = 0; tn < 4; ++tn) {
            const int n = waveN * 64 + tn * 16 + l16;
            v8bf b0 = *reinterpret_cast<const v8bf*>(&Bb[n * LDSW + lHi * 8]);
            v8bf b1 = *reinterpret_cast<const v8bf*>(&Bb[n * LDSW + 16 + lHi * 8]);
            bfrag[tn] = __builtin_shufflevector(b0, b1, 0, 1, 2, 3, 4, 5, 6, 7,
                                                8, 9, 10, 11, 12, 13, 14, 15);
        }
#pragma unroll
        for (int tm = 0; tm < 2; ++tm)
#pragma unroll
            for (int tn = 0; tn < 4; ++tn)
                acc[tm][tn] = __builtin_amdgcn_wmma_f32_16x16x32_bf16(
                    false, afrag[tm], false, bfrag[tn], (short)0, acc[tm][tn],
                    false, false);

        // ---- finish staging of next slab, then rotate buffers ----
        if (more) {
            if (MODE != MODE_LOGITS) storeBT(nxt, u0, u1);
            wait_async_all();
        }
        __syncthreads();
    }

    // ---- epilogue: D lane mapping row = g + (L>>4)*8, col = L&15 ----
    float*  oF = outF + (size_t)z * oBatch;
    bf16_t* oB = outB + (size_t)z * oBatch;
#pragma unroll
    for (int tm = 0; tm < 2; ++tm) {
#pragma unroll
        for (int tn = 0; tn < 4; ++tn) {
            const long long col = nBase + waveN * 64 + tn * 16 + l16;
#pragma unroll
            for (int g = 0; g < 8; ++g) {
                const long long row = mBase + waveM * 32 + tm * 16 + lHi * 8 + g;
                float v = acc[tm][tn][g];
                if (MODE == MODE_PROJ) {
                    v += bias[col];
                    oB[row * ldo + col] = (bf16_t)v;
                } else if (MODE == MODE_LOGITS) {
                    v *= scale;
                    if (col > row) v = -INFINITY;   // strict upper triangle (s > t)
                    oF[row * ldo + col] = v;
                } else { // MODE_AV
                    oF[row * ldo + col] = v;
                }
            }
        }
    }
}

// ---------------------------------------------------------------------------
// Column softmax over the query axis t (reference: softmax(..., axis=1)).
// One thread per column s; consecutive threads -> coalesced loads along s.
// Online max/sum from t = s (entries t < s are masked to -inf -> exp = 0).
// ---------------------------------------------------------------------------
__global__ void col_softmax(const float* __restrict__ logits,
                            bf16_t* __restrict__ attnb) {
    const int s = blockIdx.x * blockDim.x + threadIdx.x;
    const int n = blockIdx.y;
    const float* L = logits + (size_t)n * TDIM * TDIM;
    bf16_t* Ab = attnb + (size_t)n * TDIM * TDIM;

    float m = -INFINITY, sum = 0.0f;
    for (int t = s; t < TDIM; ++t) {
        float v = L[(size_t)t * TDIM + s];
        if (v > m) { sum *= __expf(m - v); m = v; }
        sum += __expf(v - m);
    }
    const float inv = 1.0f / sum;
    for (int t = 0; t < TDIM; ++t) {
        float v = (t < s) ? 0.0f : __expf(L[(size_t)t * TDIM + s] - m) * inv;
        Ab[(size_t)t * TDIM + s] = (bf16_t)v;
    }
}

// ---------------------------------------------------------------------------
// Copy x into out[:, :, 0:C] (float4 vectorized). out row stride = C+V = 1024.
// ---------------------------------------------------------------------------
__global__ void concat_x(const float* __restrict__ x, float* __restrict__ out) {
    const long long i   = ((long long)blockIdx.x * blockDim.x + threadIdx.x) * 4;
    const long long row = i >> 9;    // / CDIM
    const long long c   = i & 511;   // % CDIM
    *reinterpret_cast<float4*>(out + row * (CDIM + VDIM) + c) =
        *reinterpret_cast<const float4*>(x + i);
}

// ---------------------------------------------------------------------------
extern "C" void kernel_launch(void* const* d_in, const int* in_sizes, int n_in,
                              void* d_out, int out_size, void* d_ws, size_t ws_size,
                              hipStream_t stream) {
    (void)in_sizes; (void)n_in; (void)out_size; (void)ws_size;
    const float* x  = (const float*)d_in[0];
    const float* Wq = (const float*)d_in[1];
    const float* bq = (const float*)d_in[2];
    const float* Wk = (const float*)d_in[3];
    const float* bk = (const float*)d_in[4];
    const float* Wv = (const float*)d_in[5];
    const float* bv = (const float*)d_in[6];
    float* out = (float*)d_out;

    // ---- workspace carve-up (256B aligned regions) ----
    char* ws = (char*)d_ws;
    size_t off = 0;
    auto carve = [&](size_t bytes) -> void* {
        off = (off + 255) & ~(size_t)255;
        void* p = ws + off;
        off += bytes;
        return p;
    };
    const size_t NTC = (size_t)NB * TDIM * CDIM;     // 16,777,216
    const size_t NTK = (size_t)NB * TDIM * KDIM;
    const size_t NTT = (size_t)NB * TDIM * TDIM;

    bf16_t* xb   = (bf16_t*)carve(NTC * 2);
    bf16_t* Wqb  = (bf16_t*)carve((size_t)CDIM * KDIM * 2);
    bf16_t* Wkb  = (bf16_t*)carve((size_t)CDIM * KDIM * 2);
    bf16_t* Wvb  = (bf16_t*)carve((size_t)CDIM * VDIM * 2);
    bf16_t* Qb   = (bf16_t*)carve(NTK * 2);
    bf16_t* Kb   = (bf16_t*)carve(NTK * 2);
    bf16_t* Vb   = (bf16_t*)carve(NTK * 2);
    float*  lgts = (float*)carve(NTT * 4);
    bf16_t* attb = (bf16_t*)carve(NTT * 2);

    const dim3 blk(256);

    // 1) converts
    cvt_f32_to_bf16<<<dim3((unsigned)(NTC / 8 / 256)), blk, 0, stream>>>(x, xb, (long long)NTC);
    cvt_f32_to_bf16<<<dim3(128), blk, 0, stream>>>(Wq, Wqb, (long long)CDIM * KDIM);
    cvt_f32_to_bf16<<<dim3(128), blk, 0, stream>>>(Wk, Wkb, (long long)CDIM * KDIM);
    cvt_f32_to_bf16<<<dim3(128), blk, 0, stream>>>(Wv, Wvb, (long long)CDIM * VDIM);

    // 2) Q/K/V projections: [N*T, C] x [C, K] + bias -> bf16
    const dim3 gProj(KDIM / 128, (NB * TDIM) / 128, 1);
    gemm_wmma<MODE_PROJ><<<gProj, blk, 0, stream>>>(
        xb, CDIM, 0, Wqb, KDIM, 0, nullptr, Qb, KDIM, 0, bq, CDIM, 1.0f);
    gemm_wmma<MODE_PROJ><<<gProj, blk, 0, stream>>>(
        xb, CDIM, 0, Wkb, KDIM, 0, nullptr, Kb, KDIM, 0, bk, CDIM, 1.0f);
    gemm_wmma<MODE_PROJ><<<gProj, blk, 0, stream>>>(
        xb, CDIM, 0, Wvb, VDIM, 0, nullptr, Vb, VDIM, 0, bv, CDIM, 1.0f);

    // 3) logits[n,t,s] = (Q . K) / sqrt(K), masked -inf for s > t
    const float inv_sqrt_k = 1.0f / sqrtf((float)KDIM);
    const dim3 gLog(TDIM / 128, TDIM / 128, NB);
    gemm_wmma<MODE_LOGITS><<<gLog, blk, 0, stream>>>(
        Qb, KDIM, (long long)TDIM * KDIM, Kb, KDIM, (long long)TDIM * KDIM,
        lgts, nullptr, TDIM, (long long)TDIM * TDIM, nullptr, KDIM, inv_sqrt_k);

    // 4) softmax over the query axis (column-wise), write bf16 attn
    col_softmax<<<dim3(TDIM / 256, NB), blk, 0, stream>>>(lgts, attb);

    // 5) attn_out = attn @ V -> out[:, :, C:C+V]
    const dim3 gAV(VDIM / 128, TDIM / 128, NB);
    gemm_wmma<MODE_AV><<<gAV, blk, 0, stream>>>(
        attb, TDIM, (long long)TDIM * TDIM, Vb, VDIM, (long long)TDIM * VDIM,
        out + CDIM, nullptr, CDIM + VDIM, (long long)TDIM * (CDIM + VDIM),
        nullptr, TDIM, 1.0f);

    // 6) out[:, :, 0:C] = x
    concat_x<<<dim3((unsigned)(NTC / 4 / 256)), blk, 0, stream>>>(x, out);
}
